// SSA_19894288515722
// MI455X (gfx1250) — matrix operand order repaired and checked by reading, not compiled, not run on hardware
//
#include <hip/hip_runtime.h>
#include <hip/hip_bf16.h>

// ---------------------------------------------------------------------------
// Spiking self-attention block on MI455X (gfx1250).
//  - All contractions via v_wmma_f32_16x16x32_bf16 (wave32 WMMA).
//  - Projection GEMM double-buffers its tiles: W tiles via Tensor Data Mover
//    (tensor_load_to_lds + s_wait_tensorcnt), X tiles via
//    global_load_async_to_lds_b128 (+ s_wait_asynccnt); stage k+1 is issued
//    before computing stage k, so the DMA engines overlap the WMMA pipe.
//  - BN folded into bf16 weights; spikes {0,1} exact in bf16.
// ---------------------------------------------------------------------------

typedef __attribute__((ext_vector_type(16))) __bf16   v16bf;
typedef __attribute__((ext_vector_type(8)))  float    v8f;
typedef __attribute__((ext_vector_type(4)))  unsigned su4;
typedef __attribute__((ext_vector_type(8)))  unsigned su8;

union ABFrag { v16bf v; unsigned u[8]; };

__device__ __forceinline__ v8f wmma_bf16(v16bf a, v16bf b, v8f c) {
    return __builtin_amdgcn_wmma_f32_16x16x32_bf16(false, a, false, b,
                                                   (short)0, c, false, false);
}

#define C_DIM 256
#define N_DIM 1024
#define TB_DIM 64
#define CN (C_DIM * N_DIM)
#define TBCN (TB_DIM * CN)
#define BCN (16 * CN)    // per-timestep element count (B*C*N)
#define LW_PITCH 40      // halves; 80B rows (TDM pads 16-DW rows by 4 DW)
#define LX_PITCH 136     // halves; 272B rows (16B aligned for async b128)

// ---------------- BN-fold weight prep ----------------
__global__ void prep_w(const float* __restrict__ w, const float* __restrict__ g,
                       const float* __restrict__ b, const float* __restrict__ m,
                       const float* __restrict__ var, const float* __restrict__ extra,
                       __bf16* __restrict__ wout, float* __restrict__ bout) {
    int i = blockIdx.x * blockDim.x + threadIdx.x;
    if (i >= C_DIM * C_DIM) return;
    int o = i >> 8, c = i & 255;
    float s = g[o] * rsqrtf(var[o] + 1e-5f);
    wout[i] = (__bf16)(w[i] * s);
    if (c == 0) {
        float eb = extra ? extra[o] : 0.0f;
        bout[o] = (eb - m[o]) * s + b[o];
    }
}

__global__ void cast_x(const float* __restrict__ x, __bf16* __restrict__ y, int n) {
    int i = blockIdx.x * blockDim.x + threadIdx.x;
    if (i < n) y[i] = (__bf16)x[i];
}

// ---------------- Projection GEMM: Y[tb,o,n] = sum_c W'[o,c] X[tb,c,n] + bias[o]
// Block tile 64(M) x 128(N), 8 waves = 4(M)x2(N), wave tile 16x64.
// K = 256 in 8 steps of 32, ping-pong double-buffered TDM/async staging. ----
__global__ __launch_bounds__(256) void proj_gemm(const __bf16* __restrict__ W,
                                                 const float* __restrict__ bias,
                                                 const __bf16* __restrict__ X,
                                                 __bf16* __restrict__ Y) {
    __shared__ __bf16 lw[2][64 * LW_PITCH];    // [m][k], pitch 40 halves
    __shared__ __bf16 lx[2][32 * LX_PITCH];    // [k][n], pitch 136 halves
    const int tid  = threadIdx.x;
    const int lane = tid & 31;
    const int wv   = tid >> 5;
    const int nBlk = blockIdx.x * 128;
    const int oBlk = blockIdx.y * 64;
    const int tb   = blockIdx.z;
    const size_t xbase = (size_t)tb * CN;

    const int mT = (wv & 3) * 16;    // wave M tile
    const int nW = (wv >> 2) * 64;   // wave N group (4 x 16)

    v8f acc[4];
    #pragma unroll
    for (int i = 0; i < 4; ++i)
        #pragma unroll
        for (int e = 0; e < 8; ++e) acc[i][e] = 0.0f;

    const int sc = tid >> 3;          // X-stage c row 0..31
    const int sn = (tid & 7) * 16;    // X-stage n chunk

    const int g  = lane >> 4;
    const int ml = lane & 15;

    // Issue TDM (W tile, wave 0) + async-to-LDS (X tile) for stage s.
    auto stage = [&](int s) {
        const int k0  = s * 32;
        const int buf = s & 1;
        if (wv == 0) {   // W tile [64 x 32] via Tensor Data Mover
            const unsigned long long ga =
                (unsigned long long)(const void*)(W + (size_t)oBlk * C_DIM + k0);
            su4 g0; su8 g1;
            g0[0] = 1u;                                   // count=1, user mode
            g0[1] = (unsigned)(size_t)(const void*)&lw[buf][0];  // lds_addr
            g0[2] = (unsigned)ga;                         // global_addr lo
            g0[3] = (unsigned)(ga >> 32) | (2u << 30);    // global_addr hi | type=2
            // data_size=2B, pad_enable, pad_interval=16DW, pad_amount=4DW
            g1[0] = (1u << 16) | (1u << 20) | (3u << 22) | (3u << 25);
            g1[1] = 256u << 16;   // tensor_dim0 = 256 (bits 79:48 low half)
            g1[2] = 256u << 16;   // tensor_dim1 = 256 (bits 111:80 low half)
            g1[3] = 32u << 16;    // tile_dim0 = 32 elements (64B rows)
            g1[4] = 64u;          // tile_dim1 = 64 rows
            g1[5] = 256u;         // tensor_dim0_stride = 256 elements
            g1[6] = 0u;
            g1[7] = 0u;
            asm volatile("tensor_load_to_lds %0, %1, null, null"
                         :: "s"(g0), "s"(g1) : "memory");
        }
        // X tile [32 x 128] via async global->LDS (2 x 16B per thread)
        const __bf16* gsrc = X + xbase + (size_t)(k0 + sc) * N_DIM + nBlk + sn;
        const unsigned ldsa =
            (unsigned)(size_t)(const void*)&lx[buf][sc * LX_PITCH + sn];
        asm volatile("global_load_async_to_lds_b128 %0, %1, off"
                     :: "v"(ldsa), "v"(gsrc) : "memory");
        asm volatile("global_load_async_to_lds_b128 %0, %1, off"
                     :: "v"(ldsa + 16u), "v"(gsrc + 8) : "memory");
    };

    stage(0);   // prologue

    for (int ks = 0; ks < 8; ++ks) {
        // Per-wave: my stage-ks DMAs are done.
        asm volatile("s_wait_asynccnt 0x0" ::: "memory");
        __builtin_amdgcn_s_wait_tensorcnt(0);
        // All waves: stage-ks tiles visible; everyone done reading buf[ks^1]
        // from iteration ks-1, so stage ks+1 may overwrite it.
        __syncthreads();
        if (ks + 1 < 8) stage(ks + 1);   // overlap with compute below

        const int buf = ks & 1;
        ABFrag a;   // A 16x32: lane row m, K pairs per ISA 7.12.2 layout
        {
            const __bf16* row = &lw[buf][(mT + ml) * LW_PITCH];
            #pragma unroll
            for (int j = 0; j < 8; ++j) {
                const int k = ((j < 4) ? 0 : 16) + g * 8 + 2 * (j & 3);
                a.u[j] = *reinterpret_cast<const unsigned*>(row + k);
            }
        }
        #pragma unroll
        for (int nt = 0; nt < 4; ++nt) {
            ABFrag b;  // B 32x16 from k-major tile: n = ml, k = (j<4?0:16) + g*8 + 2(j&3) + p
            const int nIdx = nW + nt * 16 + ml;
            const unsigned short* lxs =
                reinterpret_cast<const unsigned short*>(&lx[buf][0]);
            #pragma unroll
            for (int j = 0; j < 8; ++j) {
                const int k = ((j < 4) ? 0 : 16) + g * 8 + 2 * (j & 3);
                const unsigned lo = lxs[k * LX_PITCH + nIdx];
                const unsigned hi = lxs[(k + 1) * LX_PITCH + nIdx];
                b.u[j] = lo | (hi << 16);
            }
            acc[nt] = wmma_bf16(a.v, b.v, acc[nt]);
        }
    }
    // epilogue: D layout m = g*8 + vgpr, n = ml
    #pragma unroll
    for (int nt = 0; nt < 4; ++nt)
        #pragma unroll
        for (int vi = 0; vi < 8; ++vi) {
            const int o = oBlk + mT + g * 8 + vi;
            const int n = nBlk + nW + nt * 16 + ml;
            Y[xbase + (size_t)o * N_DIM + n] = (__bf16)(acc[nt][vi] + bias[o]);
        }
}

// ---------------- LIF scan over T=4 ----------------
template <bool F32OUT>
__global__ void lif_scan(const __bf16* __restrict__ y, void* __restrict__ out, float vth) {
    int i = blockIdx.x * blockDim.x + threadIdx.x;
    if (i >= BCN) return;
    float mem = 0.0f;
    #pragma unroll
    for (int t = 0; t < 4; ++t) {
        float x = (float)y[(size_t)t * BCN + i];
        mem += (x - mem) * 0.5f;
        float s = (mem >= vth) ? 1.0f : 0.0f;
        mem *= (1.0f - s);
        if (F32OUT) ((float*)out)[(size_t)t * BCN + i] = s;
        else        ((__bf16*)out)[(size_t)t * BCN + i] = (__bf16)s;
    }
}

// ---------------- kv[tbh][f][e] = sum_n sk[h*32+e, n] * sv[h*32+f, n] -------
__global__ __launch_bounds__(128) void kv_gemm(const __bf16* __restrict__ sk,
                                               const __bf16* __restrict__ sv,
                                               float* __restrict__ kvb) {
    const int tid = threadIdx.x, lane = tid & 31, wv = tid >> 5;
    const int tbh = blockIdx.x;
    const int h = tbh & 7, tb = tbh >> 3;
    const size_t base = (size_t)tb * CN + (size_t)(h * 32) * N_DIM;
    const int e0 = (wv & 1) * 16, f0 = (wv >> 1) * 16;
    const int g = lane >> 4, ml = lane & 15;

    v8f acc;
    #pragma unroll
    for (int e = 0; e < 8; ++e) acc[e] = 0.0f;

    const __bf16* arow = sk + base + (size_t)(e0 + ml) * N_DIM;
    const __bf16* brow = sv + base + (size_t)(f0 + ml) * N_DIM;
    for (int s = 0; s < 32; ++s) {
        const int n0 = s * 32;
        __builtin_prefetch((const void*)(arow + n0 + 256), 0, 3);
        ABFrag a, b;
        #pragma unroll
        for (int j = 0; j < 8; ++j) {
            const int ka = ((j < 4) ? 0 : 16) + g * 8 + 2 * (j & 3);
            a.u[j] = *reinterpret_cast<const unsigned*>(arow + n0 + ka);
            b.u[j] = *reinterpret_cast<const unsigned*>(brow + n0 + ka);
        }
        acc = wmma_bf16(a.v, b.v, acc);
    }
    float* outb = kvb + (size_t)tbh * 1024;
    #pragma unroll
    for (int vi = 0; vi < 8; ++vi)
        outb[(f0 + ml) * 32 + (e0 + g * 8 + vi)] = acc[vi];
}

// ---------------- xa[h*32+e, n] = 0.125 * sum_f kv[f,e] * sq[h*32+f, n] -----
__global__ __launch_bounds__(256) void attn_gemm(const __bf16* __restrict__ sq,
                                                 const float* __restrict__ kvb,
                                                 __bf16* __restrict__ xa) {
    __shared__ __bf16 lkv[32 * 34];    // [e][f] = kv[f][e] (A matrix, transposed)
    __shared__ __bf16 lsq[128 * 34];   // [n][f]
    const int tid = threadIdx.x, lane = tid & 31, wv = tid >> 5;
    const int tbh = blockIdx.z;
    const int h = tbh & 7, tb = tbh >> 3;
    const int nBlk = blockIdx.x * 128;
    const size_t base = (size_t)tb * CN + (size_t)(h * 32) * N_DIM;

    const float* kvsrc = kvb + (size_t)tbh * 1024;
    #pragma unroll
    for (int r = 0; r < 4; ++r) {   // 1024 f32 -> bf16, transposed
        const int i = tid + r * 256;
        lkv[(i & 31) * 34 + (i >> 5)] = (__bf16)kvsrc[i];
    }
    {
        const int sc = tid >> 3;          // f row 0..31
        const int sn = (tid & 7) * 16;    // n chunk
        const __bf16* src = sq + base + (size_t)sc * N_DIM + nBlk + sn;
        union { uint4 q[2]; __bf16 hh[16]; } t;
        t.q[0] = *reinterpret_cast<const uint4*>(src);
        t.q[1] = *reinterpret_cast<const uint4*>(src + 8);
        #pragma unroll
        for (int i = 0; i < 16; ++i) lsq[(sn + i) * 34 + sc] = t.hh[i];
    }
    __syncthreads();

    const int eT = (wv & 1) * 16;
    const int nW = (wv >> 1) * 32;
    const int g = lane >> 4, ml = lane & 15;

    ABFrag a;
    {
        const __bf16* row = &lkv[(eT + ml) * 34];
        #pragma unroll
        for (int j = 0; j < 8; ++j) {
            const int k = ((j < 4) ? 0 : 16) + g * 8 + 2 * (j & 3);
            a.u[j] = *reinterpret_cast<const unsigned*>(row + k);
        }
    }
    #pragma unroll
    for (int nt = 0; nt < 2; ++nt) {
        ABFrag b;
        const __bf16* row = &lsq[(nW + nt * 16 + ml) * 34];
        #pragma unroll
        for (int j = 0; j < 8; ++j)
            b.u[j] = *reinterpret_cast<const unsigned*>(row + ((j < 4) ? 0 : 16) + g * 8 + 2 * (j & 3));
        v8f acc;
        #pragma unroll
        for (int e = 0; e < 8; ++e) acc[e] = 0.0f;
        acc = wmma_bf16(a.v, b.v, acc);
        #pragma unroll
        for (int vi = 0; vi < 8; ++vi) {
            const int c = h * 32 + eT + g * 8 + vi;
            const int n = nBlk + nW + nt * 16 + ml;
            xa[(size_t)tb * CN + (size_t)c * N_DIM + n] = (__bf16)(0.125f * acc[vi]);
        }
    }
}

// ---------------- v output: v[t,b,h,n,e] = sv[t,b, h*32+e, n] as f32 --------
__global__ void v_out(const __bf16* __restrict__ sv, float* __restrict__ out, int n) {
    int j = blockIdx.x * blockDim.x + threadIdx.x;
    if (j >= n) return;
    const int e = j & 31;
    const int nn = (j >> 5) & 1023;
    const int h = (j >> 15) & 7;
    const int b = (j >> 18) & 15;
    const int t = j >> 22;
    out[j] = (float)sv[(size_t)((t * 16 + b) * 256 + h * 32 + e) * N_DIM + nn];
}

// ---------------------------------------------------------------------------
extern "C" void kernel_launch(void* const* d_in, const int* in_sizes, int n_in,
                              void* d_out, int out_size, void* d_ws, size_t ws_size,
                              hipStream_t stream) {
    (void)in_sizes; (void)n_in; (void)out_size; (void)ws_size;
    char* p = (char*)d_ws;
    auto alloc = [&](size_t bytes) -> char* {
        char* r = p; p += (bytes + 255) & ~(size_t)255; return r;
    };
    __bf16* xbf  = (__bf16*)alloc((size_t)TBCN * 2);
    __bf16* Wf[4]; float* bf[4];
    for (int i = 0; i < 4; ++i) {
        Wf[i] = (__bf16*)alloc((size_t)C_DIM * C_DIM * 2);
        bf[i] = (float*)alloc((size_t)C_DIM * 4);
    }
    __bf16* Ybuf = (__bf16*)alloc((size_t)TBCN * 2);
    __bf16* sq   = (__bf16*)alloc((size_t)TBCN * 2);
    __bf16* skb  = (__bf16*)alloc((size_t)TBCN * 2);
    __bf16* svb  = (__bf16*)alloc((size_t)TBCN * 2);
    float*  kvb  = (float*)alloc((size_t)512 * 1024 * 4);
    __bf16* sxa  = (__bf16*)alloc((size_t)TBCN * 2);

    const float* fin[23];
    for (int i = 0; i < 23; ++i) fin[i] = (const float*)d_in[i];

    // BN-fold weights: q(2..6), k(7..11), v(12..16), p(w=17,b=18,g=19,be=20,m=21,v=22)
    prep_w<<<256, 256, 0, stream>>>(fin[2],  fin[3],  fin[4],  fin[5],  fin[6],  nullptr, Wf[0], bf[0]);
    prep_w<<<256, 256, 0, stream>>>(fin[7],  fin[8],  fin[9],  fin[10], fin[11], nullptr, Wf[1], bf[1]);
    prep_w<<<256, 256, 0, stream>>>(fin[12], fin[13], fin[14], fin[15], fin[16], nullptr, Wf[2], bf[2]);
    prep_w<<<256, 256, 0, stream>>>(fin[17], fin[19], fin[20], fin[21], fin[22], fin[18], Wf[3], bf[3]);

    cast_x<<<TBCN / 256, 256, 0, stream>>>(fin[0], xbf, TBCN);

    dim3 gproj(8, 4, 64);   // N/128, C/64, TB
    __bf16* spikes[3] = {sq, skb, svb};
    for (int r = 0; r < 3; ++r) {
        proj_gemm<<<gproj, 256, 0, stream>>>(Wf[r], bf[r], xbf, Ybuf);
        lif_scan<false><<<BCN / 256, 256, 0, stream>>>(Ybuf, spikes[r], 1.0f);
    }

    kv_gemm<<<512, 128, 0, stream>>>(skb, svb, kvb);
    attn_gemm<<<dim3(8, 1, 512), 256, 0, stream>>>(sq, kvb, Ybuf);
    lif_scan<false><<<BCN / 256, 256, 0, stream>>>(Ybuf, sxa, 0.5f);

    proj_gemm<<<gproj, 256, 0, stream>>>(Wf[3], bf[3], sxa, Ybuf);
    lif_scan<true><<<BCN / 256, 256, 0, stream>>>(Ybuf, d_out, 1.0f);

    v_out<<<TBCN / 256, 256, 0, stream>>>(svb, (float*)d_out + TBCN, TBCN);
}
